// MultiHeadAttention_45208825758014
// MI455X (gfx1250) — matrix operand order repaired and checked by reading, not compiled, not run on hardware
//
#include <hip/hip_runtime.h>
#include <cstdint>
#include <cstddef>

// ---------------------------------------------------------------------------
// Types for CDNA5 WMMA (wave32)
// ---------------------------------------------------------------------------
typedef __attribute__((ext_vector_type(16))) __bf16 v16bf;
typedef __attribute__((ext_vector_type(8)))  __bf16 v8bf;
typedef __attribute__((ext_vector_type(8)))  float  v8f;
typedef __attribute__((ext_vector_type(4)))  unsigned tdm_u4;
typedef __attribute__((ext_vector_type(8)))  int      tdm_i8;
typedef __attribute__((ext_vector_type(4)))  int      tdm_i4;

#define NHEAD 16
#define SEQ   2048
#define DIM   1024
#define HDIM  64
#define BATCH 2
#define KCHUNK 128          // K-chunk staged into LDS per TDM transfer

#if defined(__has_builtin)
#  if __has_builtin(__builtin_amdgcn_tensor_load_to_lds)
#    define HAVE_TDM 1
#  endif
#endif
#ifndef HAVE_TDM
#  define HAVE_TDM 0
#endif

__device__ __forceinline__ __bf16 f2bf(float f) {
    unsigned u = __builtin_bit_cast(unsigned, f);
    u += 0x7fffu + ((u >> 16) & 1u);              // round-to-nearest-even
    unsigned short s = (unsigned short)(u >> 16);
    return __builtin_bit_cast(__bf16, s);
}

__device__ __forceinline__ v16bf combine8(v8bf lo, v8bf hi) {
    return __builtin_shufflevector(lo, hi, 0,1,2,3,4,5,6,7,8,9,10,11,12,13,14,15);
}

__device__ __forceinline__ v8f wmma_bf16(v16bf a, v16bf b, v8f c) {
    // D = A(16x32 bf16) * B(32x16 bf16) + C(16x16 f32)
    return __builtin_amdgcn_wmma_f32_16x16x32_bf16(
        false, a, false, b, (short)0, c, false, false);
}

// A-fragment (16x32, 16-bit): lane holds row m = lane%16; element e maps to
// K = 16*(lane/16) + e  -> two contiguous 8-element (16B) runs.
__device__ __forceinline__ v16bf load_afrag(const __bf16* row, int half) {
    v8bf lo = *(const v8bf*)(row + 16 * half);
    v8bf hi = *(const v8bf*)(row + 16 * half + 8);
    return combine8(lo, hi);
}

__device__ __forceinline__ int rfl(unsigned v) {
    return __builtin_amdgcn_readfirstlane((int)v);
}

#if HAVE_TDM
// Tensor Data Mover: 2D tile load, global -> LDS.
// tile: tile_n rows x tile_k elements (bf16), row stride = stride_elts.
// D# group0: count=1 | lds_addr | global_addr | type=2
// D# group1: data_size=2B | tensor dims = tile dims (no OOB) | dim0 stride
__device__ __forceinline__ void tdm_load_2d(const void* gptr, unsigned lds_off,
                                            int tile_k, int tile_n,
                                            int stride_elts) {
    unsigned long long ga = (unsigned long long)(uintptr_t)gptr;
    tdm_u4 g0;
    g0[0] = (unsigned)rfl(1u);                                   // count=1
    g0[1] = (unsigned)rfl(lds_off);                              // lds_addr
    g0[2] = (unsigned)rfl((unsigned)ga);                         // gaddr lo
    g0[3] = (unsigned)rfl((unsigned)((ga >> 32) & 0x01FFFFFFull)
                          | (2u << 30));                         // gaddr hi|type
    tdm_i8 g1;
    g1[0] = rfl(1u << 16);                                       // data_size=2B
    g1[1] = rfl(((unsigned)tile_k & 0xFFFFu) << 16);             // tensor_dim0 lo
    g1[2] = rfl(((unsigned)tile_k >> 16)
                | (((unsigned)tile_n & 0xFFFFu) << 16));         // td0 hi|td1 lo
    g1[3] = rfl(((unsigned)tile_n >> 16)
                | (((unsigned)tile_k & 0xFFFFu) << 16));         // td1 hi|tile_dim0
    g1[4] = rfl((unsigned)tile_n & 0xFFFFu);                     // tile_dim1, tile_dim2=0
    g1[5] = rfl((unsigned)stride_elts);                          // dim0 stride lo
    g1[6] = rfl(0u);
    g1[7] = rfl(0u);
    tdm_i4 z4 = {};
#if __clang_major__ >= 23
    tdm_i8 z8 = {};
    __builtin_amdgcn_tensor_load_to_lds(g0, g1, z4, z4, z8, 0);
#else
    __builtin_amdgcn_tensor_load_to_lds(g0, g1, z4, z4, 0);
#endif
}
#endif

// ---------------------------------------------------------------------------
// Prep kernels
// ---------------------------------------------------------------------------
__global__ void cvt_f32_bf16(const float* __restrict__ in,
                             unsigned short* __restrict__ out_us) {
    __bf16* out = reinterpret_cast<__bf16*>(out_us);
    int i = blockIdx.x * blockDim.x + threadIdx.x;
    out[i] = f2bf(in[i]);
}

// out[n][k] = in[k][n], fp32 -> bf16, DIM x DIM
__global__ void transpose_cvt(const float* __restrict__ in,
                              unsigned short* __restrict__ out_us) {
    __bf16* out = reinterpret_cast<__bf16*>(out_us);
    __shared__ float tile[32][33];
    int bx = blockIdx.x * 32, by = blockIdx.y * 32;
    int tx = threadIdx.x & 31, ty = threadIdx.x >> 5;   // 32x8
    #pragma unroll
    for (int i = ty; i < 32; i += 8)
        tile[i][tx] = in[(by + i) * DIM + bx + tx];
    __syncthreads();
    #pragma unroll
    for (int i = ty; i < 32; i += 8)
        out[(bx + i) * DIM + by + tx] = f2bf(tile[tx][i]);
}

// ---------------------------------------------------------------------------
// WMMA GEMM: C[M x DIM] = A[M x DIM] @ Bt^T + bias   (Bt is [N][K] bf16)
// mode 0: out bf16, head-split [b,h,s,hd]
// mode 2: out bf16, transposed head-split [b,h,hd,s]   (for V)
// mode 3: out fp32 = acc + bias + resid  (row-major [M x DIM])
// block: 256 thr = 8 waves; wave -> 32(M) x 64(N); block -> 256 x 64.
// B tile [64 N][KCHUNK K] staged into LDS by the Tensor Data Mover.
// ---------------------------------------------------------------------------
__global__ void gemm_wmma(const unsigned short* __restrict__ A_us,
                          const unsigned short* __restrict__ Bt_us,
                          const float* __restrict__ bias,
                          const float* __restrict__ resid,
                          void* __restrict__ out, int mode) {
    const __bf16* A = reinterpret_cast<const __bf16*>(A_us);

    __shared__ alignas(64) __bf16 btile[64 * KCHUNK];   // 16 KB

    int lane = threadIdx.x & 31;
    int wid  = threadIdx.x >> 5;
    int half = lane >> 4;
    int l16  = lane & 15;
    int mbase = blockIdx.y * 256 + wid * 32;
    int nbase = blockIdx.x * 64;

    const __bf16* Arow0 = A + (size_t)(mbase + l16) * DIM;
    const __bf16* Arow1 = A + (size_t)(mbase + 16 + l16) * DIM;

    v8f acc[2][4] = {};
    unsigned lds_off = (unsigned)(uintptr_t)(&btile[0]);

    for (int ko = 0; ko < DIM; ko += KCHUNK) {
        // ---- stage B tile into LDS ----
#if HAVE_TDM
        if (threadIdx.x < 32) {                 // wave 0 drives the TDM
            tdm_load_2d(Bt_us + (size_t)nbase * DIM + ko, lds_off,
                        KCHUNK, 64, DIM);
            __builtin_amdgcn_s_wait_tensorcnt(0);
        }
#else
        {   // cooperative fallback copy (16B per thread-iteration)
            const __bf16* src = reinterpret_cast<const __bf16*>(Bt_us);
            for (int v = threadIdx.x; v < 64 * (KCHUNK / 8); v += 256) {
                int row = v >> 4, col = (v & 15) * 8;
                *(v8bf*)(btile + row * KCHUNK + col) =
                    *(const v8bf*)(src + (size_t)(nbase + row) * DIM + ko + col);
            }
        }
#endif
        __syncthreads();

        #pragma unroll
        for (int k2 = 0; k2 < KCHUNK; k2 += 32) {
            v16bf a0 = load_afrag(Arow0 + ko + k2, half);
            v16bf a1 = load_afrag(Arow1 + ko + k2, half);
            #pragma unroll
            for (int nt = 0; nt < 4; ++nt) {
                const __bf16* bp = btile + (nt * 16 + l16) * KCHUNK
                                   + k2 + 16 * half;
                v16bf b = *(const v16bf*)bp;           // 32B LDS K-run
                acc[0][nt] = wmma_bf16(a0, b, acc[0][nt]);
                acc[1][nt] = wmma_bf16(a1, b, acc[1][nt]);
            }
        }
        __syncthreads();
    }

    // Epilogue: C/D layout -> lane l16 = col, VGPR r -> row r + 8*half
    #pragma unroll
    for (int mi = 0; mi < 2; ++mi) {
        #pragma unroll
        for (int nt = 0; nt < 4; ++nt) {
            int n = nbase + nt * 16 + l16;
            float bval = bias[n];
            #pragma unroll
            for (int r = 0; r < 8; ++r) {
                int m = mbase + mi * 16 + r + 8 * half;
                float val = acc[mi][nt][r] + bval;
                if (mode == 3) {
                    ((float*)out)[(size_t)m * DIM + n] =
                        val + resid[(size_t)m * DIM + n];
                } else {
                    int b = m >> 11, s = m & (SEQ - 1);
                    int h = n >> 6,  d = n & (HDIM - 1);
                    __bf16* o = (__bf16*)out;
                    if (mode == 0)       // [b,h,s,hd]
                        o[(((size_t)(b * NHEAD + h) * SEQ + s) << 6) + d] = f2bf(val);
                    else                 // mode 2: [b,h,hd,s]
                        o[(((size_t)(b * NHEAD + h) << 6) + d) * SEQ + s] = f2bf(val);
                }
            }
        }
    }
}

// ---------------------------------------------------------------------------
// Flash attention: per wave one 16-row q-tile, online softmax over 32-wide
// key tiles; score + PV matmuls via WMMA.  grid = B*H*(S/128), block = 256.
// ---------------------------------------------------------------------------
__global__ void attn_wmma(const unsigned short* __restrict__ Q_us,
                          const unsigned short* __restrict__ K_us,
                          const unsigned short* __restrict__ Vt_us,
                          unsigned short* __restrict__ ctx_us) {
    const __bf16* Qb = reinterpret_cast<const __bf16*>(Q_us);
    const __bf16* Kb = reinterpret_cast<const __bf16*>(K_us);
    const __bf16* Vt = reinterpret_cast<const __bf16*>(Vt_us);
    __bf16* ctx = reinterpret_cast<__bf16*>(ctx_us);

    int lane = threadIdx.x & 31;
    int wid  = threadIdx.x >> 5;
    int half = lane >> 4;
    int l16  = lane & 15;

    int bh = blockIdx.x >> 4;            // b*NHEAD + h
    int qt = blockIdx.x & 15;            // q-tile-of-128 index
    int b  = bh >> 4, h = bh & 15;
    int qrow = qt * 128 + wid * 16;

    // Q A-fragments (K-dim = hd = 64 -> two 16x32 fragments)
    const __bf16* qptr = Qb + ((size_t)bh * SEQ + qrow + l16) * HDIM;
    v16bf a0 = load_afrag(qptr,      half);
    v16bf a1 = load_afrag(qptr + 32, half);

    const __bf16* kbase = Kb + (size_t)bh * SEQ * HDIM;
    const __bf16* vbase = Vt + (size_t)bh * HDIM * SEQ;

    v8f accc[4] = {};
    float mrow[8], lrow[8];
    #pragma unroll
    for (int r = 0; r < 8; ++r) { mrow[r] = -3.0e38f; lrow[r] = 0.f; }

    __shared__ alignas(32) __bf16 pT[8][16 * 32];    // per-wave P tile
    __bf16* myP = &pT[wid][0];

    for (int j = 0; j < SEQ; j += 32) {
        if (j + 32 < SEQ)   // speculative prefetch of next K tile
            __builtin_prefetch(kbase + (size_t)(j + 32 + l16) * HDIM, 0, 0);

        // ---- scores: S = Q @ K^T * 1/sqrt(hd) ----
        v8f s0 = {}, s1 = {};
        {
            const __bf16* kp0 = kbase + (size_t)(j + l16) * HDIM + 16 * half;
            s0 = wmma_bf16(a0, *(const v16bf*)kp0,        s0);
            s0 = wmma_bf16(a1, *(const v16bf*)(kp0 + 32), s0);
            const __bf16* kp1 = kbase + (size_t)(j + 16 + l16) * HDIM + 16 * half;
            s1 = wmma_bf16(a0, *(const v16bf*)kp1,        s1);
            s1 = wmma_bf16(a1, *(const v16bf*)(kp1 + 32), s1);
        }

        // ---- online softmax (row r lives in 16-lane half; reduce via shfl) ----
        float alpha[8];
        #pragma unroll
        for (int r = 0; r < 8; ++r) {
            float v0 = s0[r] * 0.125f;               // 1/sqrt(64)
            float v1 = s1[r] * 0.125f;
            float t = fmaxf(v0, v1);
            t = fmaxf(t, __shfl_xor(t, 1, 32));
            t = fmaxf(t, __shfl_xor(t, 2, 32));
            t = fmaxf(t, __shfl_xor(t, 4, 32));
            t = fmaxf(t, __shfl_xor(t, 8, 32));
            float mnew = fmaxf(mrow[r], t);
            float al   = __expf(mrow[r] - mnew);
            mrow[r] = mnew;
            float p0 = __expf(v0 - mnew);
            float p1 = __expf(v1 - mnew);
            s0[r] = p0; s1[r] = p1;
            float rs = p0 + p1;
            rs += __shfl_xor(rs, 1, 32);
            rs += __shfl_xor(rs, 2, 32);
            rs += __shfl_xor(rs, 4, 32);
            rs += __shfl_xor(rs, 8, 32);
            lrow[r] = lrow[r] * al + rs;
            alpha[r] = al;
        }
        #pragma unroll
        for (int nt = 0; nt < 4; ++nt)
            #pragma unroll
            for (int r = 0; r < 8; ++r)
                accc[nt][r] = accc[nt][r] * alpha[r];

        // ---- P (C-layout) -> LDS -> A-fragment ----
        #pragma unroll
        for (int r = 0; r < 8; ++r) {
            int row = r + 8 * half;
            myP[row * 32 + l16]      = f2bf(s0[r]);
            myP[row * 32 + 16 + l16] = f2bf(s1[r]);
        }
        asm volatile("s_wait_dscnt 0" ::: "memory");  // same-wave DS RAW fence
        v16bf ap = load_afrag(myP + l16 * 32, half);

        // ---- ctx += P @ V  (V stored [hd][s]: contiguous K runs) ----
        #pragma unroll
        for (int nt = 0; nt < 4; ++nt) {
            const __bf16* vp = vbase + (size_t)(nt * 16 + l16) * SEQ + j + 16 * half;
            accc[nt] = wmma_bf16(ap, *(const v16bf*)vp, accc[nt]);
        }
    }

    // ---- finalize: divide by row sums, store bf16 ctx [b, s, h*64+n] ----
    #pragma unroll
    for (int nt = 0; nt < 4; ++nt) {
        int n = nt * 16 + l16;
        #pragma unroll
        for (int r = 0; r < 8; ++r) {
            int srow = qrow + r + 8 * half;
            float val = accc[nt][r] / lrow[r];
            ctx[((size_t)(b * SEQ + srow)) * DIM + h * HDIM + n] = f2bf(val);
        }
    }
}

// ---------------------------------------------------------------------------
// Row LayerNorm in place over d_out (4096 rows x 1024), eps = 1e-12
// ---------------------------------------------------------------------------
__global__ void ln_kernel(float* __restrict__ hb,
                          const float* __restrict__ gamma,
                          const float* __restrict__ beta) {
    int row = blockIdx.x, tid = threadIdx.x;
    __shared__ float red[256];
    float v[4];
    float s = 0.f;
    #pragma unroll
    for (int j = 0; j < 4; ++j) {
        v[j] = hb[(size_t)row * DIM + tid + j * 256];
        s += v[j];
    }
    red[tid] = s; __syncthreads();
    for (int off = 128; off > 0; off >>= 1) {
        if (tid < off) red[tid] += red[tid + off];
        __syncthreads();
    }
    float mean = red[0] * (1.f / DIM);
    __syncthreads();
    float vs = 0.f;
    #pragma unroll
    for (int j = 0; j < 4; ++j) { float d = v[j] - mean; vs += d * d; }
    red[tid] = vs; __syncthreads();
    for (int off = 128; off > 0; off >>= 1) {
        if (tid < off) red[tid] += red[tid + off];
        __syncthreads();
    }
    float rstd = rsqrtf(red[0] * (1.f / DIM) + 1e-12f);
    #pragma unroll
    for (int j = 0; j < 4; ++j) {
        int col = tid + j * 256;
        hb[(size_t)row * DIM + col] = gamma[col] * (v[j] - mean) * rstd + beta[col];
    }
}

// ---------------------------------------------------------------------------
// Host launcher
// ---------------------------------------------------------------------------
extern "C" void kernel_launch(void* const* d_in, const int* in_sizes, int n_in,
                              void* d_out, int out_size, void* d_ws, size_t ws_size,
                              hipStream_t stream) {
    (void)in_sizes; (void)n_in; (void)out_size; (void)ws_size;
    const float* x     = (const float*)d_in[0];
    const float* wq    = (const float*)d_in[1];
    const float* bq    = (const float*)d_in[2];
    const float* wk    = (const float*)d_in[3];
    const float* bk    = (const float*)d_in[4];
    const float* wv    = (const float*)d_in[5];
    const float* bv    = (const float*)d_in[6];
    const float* wo    = (const float*)d_in[7];
    const float* bo    = (const float*)d_in[8];
    const float* gamma = (const float*)d_in[9];
    const float* beta  = (const float*)d_in[10];

    const size_t MB = 1u << 20;
    char* w = (char*)d_ws;
    unsigned short* xb   = (unsigned short*)(w + 0);        // 8 MB  bf16 x
    unsigned short* wqt  = (unsigned short*)(w + 8 * MB);   // 2 MB
    unsigned short* wkt  = (unsigned short*)(w + 10 * MB);  // 2 MB
    unsigned short* wvt  = (unsigned short*)(w + 12 * MB);  // 2 MB
    unsigned short* wot  = (unsigned short*)(w + 14 * MB);  // 2 MB
    unsigned short* Qb   = (unsigned short*)(w + 16 * MB);  // 8 MB [b,h,s,hd]
    unsigned short* Kb   = (unsigned short*)(w + 24 * MB);  // 8 MB [b,h,s,hd]
    unsigned short* Vt   = (unsigned short*)(w + 32 * MB);  // 8 MB [b,h,hd,s]
    unsigned short* ctxb = (unsigned short*)(w + 40 * MB);  // 8 MB [b,s,D]

    const int M = BATCH * SEQ;                 // 4096
    // 1) conversions / weight transposes
    cvt_f32_bf16<<<(M * DIM) / 256, 256, 0, stream>>>(x, xb);
    dim3 tg(DIM / 32, DIM / 32);
    transpose_cvt<<<tg, 256, 0, stream>>>(wq, wqt);
    transpose_cvt<<<tg, 256, 0, stream>>>(wk, wkt);
    transpose_cvt<<<tg, 256, 0, stream>>>(wv, wvt);
    transpose_cvt<<<tg, 256, 0, stream>>>(wo, wot);

    // 2) QKV projections (WMMA, TDM-staged B tiles)
    dim3 gg(DIM / 64, M / 256);
    gemm_wmma<<<gg, 256, 0, stream>>>(xb, wqt, bq, nullptr, Qb, 0);
    gemm_wmma<<<gg, 256, 0, stream>>>(xb, wkt, bk, nullptr, Kb, 0);
    gemm_wmma<<<gg, 256, 0, stream>>>(xb, wvt, bv, nullptr, Vt, 2);

    // 3) flash attention (WMMA)
    attn_wmma<<<BATCH * NHEAD * (SEQ / 128), 256, 0, stream>>>(Qb, Kb, Vt, ctxb);

    // 4) output projection + bias + residual -> d_out (fp32)
    gemm_wmma<<<gg, 256, 0, stream>>>(ctxb, wot, bo, x, d_out, 3);

    // 5) LayerNorm in place
    ln_kernel<<<M, 256, 0, stream>>>((float*)d_out, gamma, beta);
}
